// PeriDynamicAttentionFlex_82317343195808
// MI455X (gfx1250) — matrix operand order repaired and checked by reading, compiled, not run on hardware
//
#include <hip/hip_runtime.h>
#include <cmath>
#include <cstdint>

typedef __attribute__((ext_vector_type(16))) _Float16 v16h;
typedef __attribute__((ext_vector_type(8)))  _Float16 v8h;
typedef __attribute__((ext_vector_type(4)))  _Float16 v4h;
typedef __attribute__((ext_vector_type(2)))  _Float16 v2h;
typedef __attribute__((ext_vector_type(8)))  float    v8f;
typedef __attribute__((ext_vector_type(4)))  float    v4f;
typedef __attribute__((ext_vector_type(4)))  unsigned int v4u;
typedef __attribute__((ext_vector_type(8)))  int      v8i;
typedef __attribute__((ext_vector_type(4)))  int      v4i;

#define B_    2
#define T_    4096
#define C_    1024
#define NH_   16
#define HS_   64
#define BD_   16
#define DELTA 32
#define M_    (B_*T_)   // 8192

#if defined(__HIP_DEVICE_COMPILE__) && __has_builtin(__builtin_amdgcn_tensor_load_to_lds)
#define USE_TDM 1
#endif

// ---------------------------------------------------------------------------
// GEMM: C[M,N](f32) = A[M,K](f16) @ BT[N,K](f16)^T
// BM=128, BN=128, BK=32; 256 threads (8 waves); each wave: 2x4 16x16 tiles.
// Tiles staged to LDS either by TDM (tensor_load_to_lds) or b128 copies.
// ---------------------------------------------------------------------------
__global__ __launch_bounds__(256)
void gemm_wmma_f16(const _Float16* __restrict__ A,
                   const _Float16* __restrict__ BT,
                   float* __restrict__ C, int M, int N, int K)
{
    __shared__ _Float16 smem[8192];        // [0,4096) = sA 128x32, [4096,8192) = sB 128x32
    _Float16* sA = smem;
    _Float16* sB = smem + 4096;

    const int tid   = threadIdx.x;
    const int lane  = tid & 31;
    const int wave  = tid >> 5;
    const int waveM = wave & 3;            // 4 waves x 32 rows
    const int waveN = wave >> 2;           // 2 waves x 64 cols
    const int hi    = lane >> 4;
    const int lm    = lane & 15;
    const int mBlock = blockIdx.x * 128;
    const int nBlock = blockIdx.y * 128;

    v8f acc[2][4] = {};

    for (int kt = 0; kt < K; kt += 32) {
#ifdef USE_TDM
        if (wave == 0) {
            // ---- D# for A tile: 2D tensor [M][K] f16, tile 128x32 ------
            unsigned long long gA = (unsigned long long)(uintptr_t)A +
                ((unsigned long long)mBlock * (unsigned)K + (unsigned)kt) * 2ull;
            v4u g0a = { 1u, 0u, (unsigned)gA,
                        (unsigned)((gA >> 32) & 0x1FFFFFFull) | (2u << 30) };
            v8i g1a = { (int)(1u << 16),                                        // data_size=2B
                        (int)(((unsigned)K & 0xFFFFu) << 16),                   // tdim0 lo
                        (int)(((unsigned)K >> 16) | (((unsigned)M & 0xFFFFu) << 16)),
                        (int)(((unsigned)M >> 16) | (32u << 16)),               // tile_dim0=32
                        128,                                                    // tile_dim1=128
                        K, 0, 0 };                                              // dim0_stride=K
            // ---- D# for B tile: 2D tensor [N][K] f16, tile 128x32 ------
            unsigned long long gB = (unsigned long long)(uintptr_t)BT +
                ((unsigned long long)nBlock * (unsigned)K + (unsigned)kt) * 2ull;
            v4u g0b = { 1u, 8192u, (unsigned)gB,
                        (unsigned)((gB >> 32) & 0x1FFFFFFull) | (2u << 30) };
            v8i g1b = { (int)(1u << 16),
                        (int)(((unsigned)K & 0xFFFFu) << 16),
                        (int)(((unsigned)K >> 16) | (((unsigned)N & 0xFFFFu) << 16)),
                        (int)(((unsigned)N >> 16) | (32u << 16)),
                        128,
                        K, 0, 0 };
            v4i z4 = { 0, 0, 0, 0 };
#if defined(__clang_major__) && __clang_major__ >= 23
            v8i z8 = { 0, 0, 0, 0, 0, 0, 0, 0 };
            __builtin_amdgcn_tensor_load_to_lds(g0a, g1a, z4, z4, z8, 0);
            __builtin_amdgcn_tensor_load_to_lds(g0b, g1b, z4, z4, z8, 0);
#else
            __builtin_amdgcn_tensor_load_to_lds(g0a, g1a, z4, z4, 0);
            __builtin_amdgcn_tensor_load_to_lds(g0b, g1b, z4, z4, 0);
#endif
            __builtin_amdgcn_s_wait_tensorcnt(0);
        }
#else
        // ---- manual staging: pure b128 copies (f16 both sides) ---------
        #pragma unroll
        for (int u = 0; u < 2; ++u) {
            int q = tid + 256 * u;                 // 512 chunks of 8 halfs
            int r = q >> 2, c8 = q & 3;
            *(v8h*)&sA[r * 32 + c8 * 8] =
                *(const v8h*)(A + (size_t)(mBlock + r) * K + kt + c8 * 8);
        }
        #pragma unroll
        for (int u = 0; u < 2; ++u) {
            int q = tid + 256 * u;
            int r = q >> 2, c8 = q & 3;
            *(v8h*)&sB[r * 32 + c8 * 8] =
                *(const v8h*)(BT + (size_t)(nBlock + r) * K + kt + c8 * 8);
        }
#endif
        __syncthreads();

        // ---- fragments per gfx1250 WMMA VGPR layouts -------------------
        v16h afrag[2], bfrag[4];
        #pragma unroll
        for (int i = 0; i < 2; ++i) {
            int m = (waveM * 32 + i * 16 + lm) * 32;
            #pragma unroll
            for (int v = 0; v < 8; ++v) {
                int kb = (v < 4) ? (hi * 8 + v * 2) : (16 + hi * 8 + (v - 4) * 2);
                v2h p = *(const v2h*)&sA[m + kb];
                afrag[i][2 * v]     = p[0];
                afrag[i][2 * v + 1] = p[1];
            }
        }
        #pragma unroll
        for (int j = 0; j < 4; ++j) {
            int n = (waveN * 64 + j * 16 + lm) * 32;
            #pragma unroll
            for (int v = 0; v < 8; ++v) {
                int kb = (v < 4) ? (hi * 8 + v * 2) : (16 + hi * 8 + (v - 4) * 2);
                v2h p = *(const v2h*)&sB[n + kb];
                bfrag[j][2 * v]     = p[0];
                bfrag[j][2 * v + 1] = p[1];
            }
        }
        #pragma unroll
        for (int i = 0; i < 2; ++i)
            #pragma unroll
            for (int j = 0; j < 4; ++j)
                acc[i][j] = __builtin_amdgcn_wmma_f32_16x16x32_f16(
                    false, afrag[i], false, bfrag[j],
                    (short)0, acc[i][j], false, false);
        __syncthreads();
    }

    #pragma unroll
    for (int i = 0; i < 2; ++i)
        #pragma unroll
        for (int j = 0; j < 4; ++j) {
            int nCol = nBlock + waveN * 64 + j * 16 + lm;
            #pragma unroll
            for (int v = 0; v < 8; ++v) {
                int mRow = mBlock + waveM * 32 + i * 16 + v + 8 * hi;
                C[(size_t)mRow * N + nCol] = acc[i][j][v];
            }
        }
}

// ---------------------------------------------------------------------------
// x (f32) -> xh (f16), vectorized x4
// ---------------------------------------------------------------------------
__global__ void pack_f16_kernel(const float* __restrict__ in,
                                _Float16* __restrict__ out, int n4)
{
    int idx = blockIdx.x * 256 + threadIdx.x;
    if (idx >= n4) return;
    v4f a = *(const v4f*)(in + (size_t)idx * 4);
    v4h h;
    #pragma unroll
    for (int i = 0; i < 4; ++i) h[i] = (_Float16)a[i];
    *(v4h*)(out + (size_t)idx * 4) = h;
}

// W2T[h*32+o][c] = sum_d W_disp[c, h*16+d] * W_sf[d, o]   (f16, [512][1024])
__global__ void prep_w2t_kernel(const float* __restrict__ Wd,
                                const float* __restrict__ Wsf,
                                _Float16* __restrict__ W2T)
{
    int idx = blockIdx.x * 256 + threadIdx.x;   // 512*1024
    if (idx >= 512 * C_) return;
    int c    = idx & (C_ - 1);
    int nCol = idx >> 10;
    int o    = nCol & 31;
    int h    = nCol >> 5;
    float s = 0.f;
    #pragma unroll
    for (int d = 0; d < 16; ++d)
        s += Wd[c * 256 + h * 16 + d] * Wsf[d * 32 + o];
    W2T[idx] = (_Float16)s;
}

// LDS-tiled transpose+pack: out[N][K] f16 = in[K][N] f32 (K=N=1024)
__global__ __launch_bounds__(256)
void transpose_pack_kernel(const float* __restrict__ in,
                           _Float16* __restrict__ out)
{
    __shared__ float tile[32][33];
    int tx = threadIdx.x & 31;
    int ty = threadIdx.x >> 5;       // 0..7
    int kb = blockIdx.x * 32;
    int nb = blockIdx.y * 32;
    #pragma unroll
    for (int u = 0; u < 4; ++u) {
        int r = ty + u * 8;
        tile[r][tx] = in[(size_t)(kb + r) * C_ + nb + tx];
    }
    __syncthreads();
    #pragma unroll
    for (int u = 0; u < 4; ++u) {
        int r = ty + u * 8;
        out[(size_t)(nb + r) * C_ + kb + tx] = (_Float16)tile[tx][r];
    }
}

// pos_feat[j, o] = sum_d rel_pos_emb[j, d] * W_pos[d, o]
__global__ void prep_pf_kernel(const float* __restrict__ rel,
                               const float* __restrict__ Wp,
                               float* __restrict__ pf)
{
    int idx = blockIdx.x * 256 + threadIdx.x;   // 32*16
    if (idx >= DELTA * BD_) return;
    int o = idx & 15;
    int j = idx >> 4;
    float s = 0.f;
    #pragma unroll
    for (int d = 0; d < 16; ++d)
        s += rel[j * 16 + d] * Wp[d * 16 + o];
    pf[idx] = s;
}

__device__ __forceinline__ float gelu_exact(float x) {
    return 0.5f * x * (1.f + erff(x * 0.70710678118654752f));
}

// ---------------------------------------------------------------------------
// Windowed attention: one wave per (b, n, t), lane = window slot j.
// fused[j] = G[t-31+j] - G[t] + b_sf  (lane 31's row IS G[t] -> shfl)
// Output written as f16 (feeds the final WMMA GEMM).
// ---------------------------------------------------------------------------
__global__ __launch_bounds__(256)
void attn_window_kernel(const float* __restrict__ G,
                        const float* __restrict__ val,
                        const float* __restrict__ pf,
                        const float* __restrict__ b_sf,
                        const float* __restrict__ w_bond,
                        const float* __restrict__ w_dmg,
                        const float* __restrict__ b_dmg,
                        _Float16* __restrict__ atth)
{
    const int lane = threadIdx.x & 31;
    const int wave = threadIdx.x >> 5;
    const long w  = (long)blockIdx.x * 8 + wave;     // (b*NH + n)*T + t
    const int t   = (int)(w % T_);
    const int bn  = (int)(w / T_);
    const int n   = bn & (NH_ - 1);
    const int b   = bn >> 4;

    const int j      = lane;
    const int src    = t - (DELTA - 1) + j;
    const bool valid = (src >= 0);
    const int srcC   = valid ? src : 0;

    const float* Grow = G + ((size_t)b * T_ + srcC) * (NH_ * 2 * BD_) + n * 32;

    float s1 = 0.f, s2 = 0.f;
    #pragma unroll
    for (int o = 0; o < 16; ++o) {
        float gj = valid ? Grow[o] : 0.f;
        float gt = __shfl(gj, 31, 32);               // lane 31 holds G[t]
        float f  = gj - gt + b_sf[o] + pf[j * 16 + o];
        s1 += gelu_exact(f) * w_bond[o];
    }
    #pragma unroll
    for (int o = 16; o < 32; ++o) {
        float gj = valid ? Grow[o] : 0.f;
        float gt = __shfl(gj, 31, 32);
        float f  = gj - gt + b_sf[o];
        s2 += gelu_exact(f) * w_dmg[o - 16];
    }
    float dmg   = 1.f / (1.f + expf(-(s2 + b_dmg[0])));
    float score = s1 - 10.f * dmg;
    if (!valid) score = -INFINITY;

    // softmax across the 32-lane wave (lane 31 always valid -> finite max)
    float mx = score;
    #pragma unroll
    for (int off = 16; off; off >>= 1)
        mx = fmaxf(mx, __shfl_xor(mx, off, 32));
    float e = valid ? expf(score - mx) : 0.f;
    float sm = e;
    #pragma unroll
    for (int off = 16; off; off >>= 1)
        sm += __shfl_xor(sm, off, 32);
    float wgt = e / sm;

    // weighted sum of val window: each lane covers h = lane and lane+32
    float acc0 = 0.f, acc1 = 0.f;
    const size_t vbase = (size_t)b * T_ * C_ + (size_t)n * HS_;
    #pragma unroll 8
    for (int jj = 0; jj < 32; ++jj) {
        float wj = __shfl(wgt, jj, 32);              // 0 for invalid slots
        int sj = t - (DELTA - 1) + jj;
        sj = sj < 0 ? 0 : sj;                        // clamped row x 0 weight
        const float* vr = val + vbase + (size_t)sj * C_;
        acc0 += wj * vr[lane];
        acc1 += wj * vr[lane + 32];
    }
    _Float16* orow = atth + ((size_t)b * T_ + t) * C_ + n * HS_;
    orow[lane]      = (_Float16)acc0;
    orow[lane + 32] = (_Float16)acc1;
}

// ---------------------------------------------------------------------------
extern "C" void kernel_launch(void* const* d_in, const int* in_sizes, int n_in,
                              void* d_out, int out_size, void* d_ws, size_t ws_size,
                              hipStream_t stream)
{
    const float* x      = (const float*)d_in[0];
    const float* W_disp = (const float*)d_in[1];
    const float* W_val  = (const float*)d_in[2];
    const float* relpe  = (const float*)d_in[3];
    const float* W_pos  = (const float*)d_in[4];
    const float* W_sf   = (const float*)d_in[5];
    const float* b_sf   = (const float*)d_in[6];
    const float* w_bond = (const float*)d_in[7];
    const float* w_dmg  = (const float*)d_in[8];
    const float* b_dmg  = (const float*)d_in[9];
    const float* W_cprj = (const float*)d_in[10];
    float* out = (float*)d_out;

    // workspace layout (float offsets; all 16B aligned)
    float* ws = (float*)d_ws;
    _Float16* xh   = (_Float16*)(ws);             // 8192*1024 f16 = 4194304 fl
    _Float16* W2T  = (_Float16*)(ws + 4194304);   // 512*1024  f16 =  262144 fl
    _Float16* WvT  = (_Float16*)(ws + 4456448);   // 1024*1024 f16 =  524288 fl
    _Float16* WcT  = (_Float16*)(ws + 4980736);   // 1024*1024 f16 =  524288 fl
    float*    pf   = ws + 5505024;                // 512 fl
    float*    G    = ws + 5505536;                // 8192*512  f32 = 4194304 fl
    float*    valb = ws + 9699840;                // 8192*1024 f32 = 8388608 fl
    _Float16* atth = (_Float16*)(ws + 18088448);  // 8192*1024 f16 = 4194304 fl

    // 1) precision/layout prep
    pack_f16_kernel<<<(M_ * C_ / 4 + 255) / 256, 256, 0, stream>>>(x, xh, M_ * C_ / 4);
    prep_w2t_kernel<<<(512 * C_ + 255) / 256, 256, 0, stream>>>(W_disp, W_sf, W2T);
    transpose_pack_kernel<<<dim3(32, 32), 256, 0, stream>>>(W_val, WvT);
    transpose_pack_kernel<<<dim3(32, 32), 256, 0, stream>>>(W_cprj, WcT);
    prep_pf_kernel<<<2, 256, 0, stream>>>(relpe, W_pos, pf);

    // 2) G = x @ W2           (8192 x 1024 x 512)
    gemm_wmma_f16<<<dim3(M_ / 128, 512 / 128), 256, 0, stream>>>(xh, W2T, G, M_, 512, C_);
    // 3) val = x @ W_val      (8192 x 1024 x 1024)
    gemm_wmma_f16<<<dim3(M_ / 128, C_ / 128), 256, 0, stream>>>(xh, WvT, valb, M_, C_, C_);
    // 4) windowed softmax-attention
    attn_window_kernel<<<(B_ * NH_ * T_) / 8, 256, 0, stream>>>(
        G, valb, pf, b_sf, w_bond, w_dmg, b_dmg, atth);
    // 5) out = att @ W_cproj  (8192 x 1024 x 1024)
    gemm_wmma_f16<<<dim3(M_ / 128, C_ / 128), 256, 0, stream>>>(atth, WcT, out, M_, C_, C_);
}
